// LensGatedAttention_65635690218094
// MI455X (gfx1250) — matrix operand order repaired and checked by reading, not compiled, not run on hardware
//
#include <hip/hip_runtime.h>

// ---------------------------------------------------------------------------
// LensGatedAttention for MI455X (gfx1250, wave32, WMMA)
//   B=1, T=4096, E=768, H=12, D=64, causal band window=512
// ---------------------------------------------------------------------------

typedef __attribute__((ext_vector_type(16))) _Float16 v16h;
typedef __attribute__((ext_vector_type(8)))  _Float16 v8h;
typedef __attribute__((ext_vector_type(4)))  _Float16 v4h;
typedef __attribute__((ext_vector_type(8)))  float    v8f;
typedef __attribute__((ext_vector_type(4)))  float    v4f;
typedef int gi4 __attribute__((vector_size(16)));   // matches builtin param type

constexpr int T    = 4096;
constexpr int E    = 768;
constexpr int H    = 12;
constexpr int HD   = 768;   // H*D
constexpr int WIN  = 512;
constexpr int SMAX = 544;   // max band columns for a 16-row query block (32-aligned)

__device__ __forceinline__ v8f wmma_f16(v16h a, v16h b, v8f c) {
  return __builtin_amdgcn_wmma_f32_16x16x32_f16(
      false, a, false, b, (short)0, c, false, false);
}

__device__ __forceinline__ v16h cat8(v8h lo, v8h hi) {
  return __builtin_shufflevector(lo, hi, 0, 1, 2, 3, 4, 5, 6, 7,
                                 8, 9, 10, 11, 12, 13, 14, 15);
}

// ---- async global->LDS copy (CDNA5 path) if the toolchain exposes it ------
#if __has_builtin(__builtin_amdgcn_global_load_async_to_lds_b128) && \
    __has_builtin(__builtin_amdgcn_s_wait_asynccnt)
#define HAVE_ASYNC_LDS 1
__device__ __forceinline__ void lds_copy16(void* l, const void* g) {
  __builtin_amdgcn_global_load_async_to_lds_b128(
      (__attribute__((address_space(1))) gi4*)(void*)g,
      (__attribute__((address_space(3))) gi4*)l, 0, 0);
}
__device__ __forceinline__ void lds_copy_wait() {
  __builtin_amdgcn_s_wait_asynccnt(0);
}
#else
__device__ __forceinline__ void lds_copy16(void* l, const void* g) {
  *(uint4*)l = *(const uint4*)g;
}
__device__ __forceinline__ void lds_copy_wait() {}
#endif

// ---------------------------------------------------------------------------
// Zero-fill the 805MB attn tensor with non-temporal stores
// ---------------------------------------------------------------------------
__global__ __launch_bounds__(256) void fill_zero_f4(float* __restrict__ p,
                                                    long long n4) {
  long long i = (long long)blockIdx.x * 256 + threadIdx.x;
  if (i < n4) {
    v4f z = {0.f, 0.f, 0.f, 0.f};
    __builtin_nontemporal_store(z, (v4f*)p + i);
  }
}

// ---------------------------------------------------------------------------
// GEMM: C[M,N] = A[M,K] * B[K,N]  (f32 in, f32 and/or f16 out, f16 WMMA)
// 128 threads = 4 waves; 64x64 C-tile; each wave a 32x32 tile = 2x2 WMMA tiles.
// Double-buffered LDS: tile t+1 is loaded to registers while WMMAs consume
// tile t; one barrier per k-iteration.
// ---------------------------------------------------------------------------
__global__ __launch_bounds__(128) void wmma_gemm64(
    const float* __restrict__ A, const float* __restrict__ B,
    float* __restrict__ C, _Float16* __restrict__ Ch,
    int M, int N, int K) {
  __shared__ alignas(32) _Float16 As[2][64 * 64];
  __shared__ alignas(32) _Float16 Bt[2][64 * 64];  // [n_local][k_local]

  const int tid  = threadIdx.x;
  const int lane = tid & 31;
  const int wave = tid >> 5;
  const int rw   = wave >> 1;
  const int cw   = wave & 1;
  const int l15  = lane & 15;
  const int hi   = (lane >> 4) & 1;
  const int hi8  = hi * 8, hi16 = hi * 16;
  const int m0   = blockIdx.y * 64;
  const int n0   = blockIdx.x * 64;

  v8f acc[2][2];
#pragma unroll
  for (int rt = 0; rt < 2; ++rt)
#pragma unroll
    for (int ct = 0; ct < 2; ++ct)
      acc[rt][ct] = (v8f){0.f, 0.f, 0.f, 0.f, 0.f, 0.f, 0.f, 0.f};

  v4f fa[8], fb[8];

  auto load_tile = [&](int kk) {
#pragma unroll
    for (int j = 0; j < 8; ++j) {      // 16 b128 loads in flight
      int i4 = tid + j * 128;
      int r = i4 >> 4, c4 = i4 & 15;
      fa[j] = *(const v4f*)&A[(size_t)(m0 + r) * K + kk + c4 * 4];
      fb[j] = *(const v4f*)&B[(size_t)(kk + r) * N + n0 + c4 * 4];
    }
  };
  auto store_tile = [&](int buf) {
#pragma unroll
    for (int j = 0; j < 8; ++j) {
      int i4 = tid + j * 128;
      int r = i4 >> 4, c4 = i4 & 15;
      v4h hv = {(_Float16)fa[j].x, (_Float16)fa[j].y,
                (_Float16)fa[j].z, (_Float16)fa[j].w};
      *(v4h*)&As[buf][r * 64 + c4 * 4] = hv;       // row-major A
      Bt[buf][(c4 * 4 + 0) * 64 + r] = (_Float16)fb[j].x;  // transposed B
      Bt[buf][(c4 * 4 + 1) * 64 + r] = (_Float16)fb[j].y;
      Bt[buf][(c4 * 4 + 2) * 64 + r] = (_Float16)fb[j].z;
      Bt[buf][(c4 * 4 + 3) * 64 + r] = (_Float16)fb[j].w;
    }
  };
  auto compute = [&](int buf) {
#pragma unroll
    for (int kt = 0; kt < 2; ++kt) {
      v16h af[2], bf[2];
#pragma unroll
      for (int rt = 0; rt < 2; ++rt) {
        const v8h* pr = (const v8h*)&As[buf][(rw * 32 + rt * 16 + l15) * 64];
        af[rt] = cat8(pr[kt * 4 + 2 * hi], pr[kt * 4 + 2 * hi + 1]);
      }
#pragma unroll
      for (int ct = 0; ct < 2; ++ct)
        bf[ct] = *(const v16h*)&Bt[buf][(cw * 32 + ct * 16 + l15) * 64 +
                                        kt * 32 + hi16];
#pragma unroll
      for (int rt = 0; rt < 2; ++rt)
#pragma unroll
        for (int ct = 0; ct < 2; ++ct)
          acc[rt][ct] = wmma_f16(af[rt], bf[ct], acc[rt][ct]);
    }
  };

  const int nTiles = K / 64;
  load_tile(0);
  store_tile(0);
  __syncthreads();
  for (int t = 0; t < nTiles; ++t) {
    if (t + 1 < nTiles) load_tile((t + 1) * 64);   // overlap with WMMAs below
    compute(t & 1);
    if (t + 1 < nTiles) {
      store_tile((t + 1) & 1);
      __syncthreads();
    }
  }

#pragma unroll
  for (int rt = 0; rt < 2; ++rt)
#pragma unroll
    for (int ct = 0; ct < 2; ++ct)
#pragma unroll
      for (int i = 0; i < 8; ++i) {
        size_t off = (size_t)(m0 + rw * 32 + rt * 16 + i + hi8) * N +
                     n0 + cw * 32 + ct * 16 + l15;
        if (C)  C[off]  = acc[rt][ct][i];
        if (Ch) Ch[off] = (_Float16)acc[rt][ct][i];
      }
}

// ---------------------------------------------------------------------------
// Per-head discoverability bias: db[h,t] = K[t]·u[h] - mean_t(K[t]·u[h])
// ---------------------------------------------------------------------------
__global__ __launch_bounds__(256) void dbias_kernel(const float* __restrict__ Km,
                                                    const float* __restrict__ u,
                                                    float* __restrict__ db) {
  const int h   = blockIdx.x;
  const int tid = threadIdx.x;
  __shared__ float us[64];
  __shared__ float red[256];
  if (tid < 64) us[tid] = u[h * 64 + tid];
  __syncthreads();

  float vals[16];
  float local = 0.f;
#pragma unroll 1
  for (int it = 0; it < 16; ++it) {
    int t = it * 256 + tid;
    const float* row = Km + (size_t)t * HD + h * 64;
    float s = 0.f;
#pragma unroll
    for (int d = 0; d < 64; ++d) s += row[d] * us[d];
    vals[it] = s;
    local += s;
  }
  red[tid] = local;
  __syncthreads();
  for (int off = 128; off > 0; off >>= 1) {
    if (tid < off) red[tid] += red[tid + off];
    __syncthreads();
  }
  const float mean = red[0] * (1.0f / (float)T);
  for (int it = 0; it < 16; ++it) {
    int t = it * 256 + tid;
    db[h * T + t] = vals[it] - mean;
  }
}

// ---------------------------------------------------------------------------
// Banded gated attention. One wave32 per (16-row query block, head).
// ---------------------------------------------------------------------------
__global__ __launch_bounds__(32) void attn_kernel(
    const _Float16* __restrict__ Qh, const _Float16* __restrict__ Kh,
    const _Float16* __restrict__ Vh, const float* __restrict__ db,
    const float* __restrict__ gates, float* __restrict__ attn,
    float* __restrict__ ctx) {
  __shared__ alignas(32) float    Ll[16 * SMAX];   // band logits (f32)
  __shared__ alignas(32) _Float16 Pl[16 * SMAX];   // band probabilities (f16)
  __shared__ alignas(32) float    dbs[SMAX];       // band bias values
  __shared__ alignas(32) _Float16 Tt[16 * 64];     // Q_t / K_t staging
  __shared__ alignas(32) _Float16 Qs[32 * 64];     // Q s-chunk (S^T term)
  __shared__ alignas(32) _Float16 KVs[32 * 64];    // K chunk; reused as Vt[d][s]

  const int h    = blockIdx.y;
  const int t0   = blockIdx.x * 16;
  const int lane = threadIdx.x;
  const int l15  = lane & 15;
  const int hi   = (lane >> 4) & 1;
  const int hi8  = hi * 8, hi16 = hi * 16;

  // 3-way gate softmax
  float g0 = gates[h * 3 + 0], g1 = gates[h * 3 + 1], g2 = gates[h * 3 + 2];
  float gm = fmaxf(g0, fmaxf(g1, g2));
  float e0 = __expf(g0 - gm), e1 = __expf(g1 - gm), e2 = __expf(g2 - gm);
  float gi = 1.f / (e0 + e1 + e2);
  const float wstd = e0 * gi, wrec = e1 * gi, wdisc = e2 * gi;
  const float scale = 0.125f;  // 1/sqrt(D)

  int sLo = t0 - WIN;
  if (sLo < 0) sLo = 0;
  sLo &= ~31;
  const int sHi = t0 + 15;
  const int nS  = sHi + 1 - sLo;

  // stage the band's bias values once
  for (int j = lane; j < nS; j += 32) dbs[j] = db[h * T + sLo + j];

  // ---- K rows t0..t0+15 -> Tt, build A-frags for the S^T term -------------
  v16h ak[2], aq[2];
#pragma unroll
  for (int j = 0; j < 4; ++j) {        // 16x64 halfs = 128 uint4
    int u = lane + j * 32;
    int r = u >> 3, q = u & 7;
    lds_copy16((uint4*)Tt + u,
               (const uint4*)(Kh + (size_t)(t0 + r) * HD + h * 64) + q);
  }
  lds_copy_wait();
  __syncthreads();
  {
    const v8h* pr = (const v8h*)&Tt[l15 * 64];
#pragma unroll
    for (int kt = 0; kt < 2; ++kt)
      ak[kt] = cat8(pr[kt * 4 + 2 * hi], pr[kt * 4 + 2 * hi + 1]);
  }
  __syncthreads();

  // ---- Q rows t0..t0+15 -> Tt, build A-frags for the S term ---------------
#pragma unroll
  for (int j = 0; j < 4; ++j) {
    int u = lane + j * 32;
    int r = u >> 3, q = u & 7;
    lds_copy16((uint4*)Tt + u,
               (const uint4*)(Qh + (size_t)(t0 + r) * HD + h * 64) + q);
  }
  lds_copy_wait();
  __syncthreads();
  {
    const v8h* pr = (const v8h*)&Tt[l15 * 64];
#pragma unroll
    for (int kt = 0; kt < 2; ++kt)
      aq[kt] = cat8(pr[kt * 4 + 2 * hi], pr[kt * 4 + 2 * hi + 1]);
  }

  // ---- logits over the band, 32 keys per chunk ----------------------------
  for (int c0 = sLo; c0 <= sHi; c0 += 32) {
    __syncthreads();
#pragma unroll
    for (int j = 0; j < 8; ++j) {      // 32x64 halfs = 256 uint4 per tile
      int u = lane + j * 32;
      int r = u >> 3, q = u & 7;
      lds_copy16((uint4*)KVs + u,
                 (const uint4*)(Kh + (size_t)(c0 + r) * HD + h * 64) + q);
      lds_copy16((uint4*)Qs + u,
                 (const uint4*)(Qh + (size_t)(c0 + r) * HD + h * 64) + q);
    }
    lds_copy_wait();
    __syncthreads();

#pragma unroll
    for (int nt = 0; nt < 2; ++nt) {
      v8f s1 = {0.f, 0.f, 0.f, 0.f, 0.f, 0.f, 0.f, 0.f};
      v8f s2 = {0.f, 0.f, 0.f, 0.f, 0.f, 0.f, 0.f, 0.f};
#pragma unroll
      for (int kt = 0; kt < 2; ++kt) {
        int base = (nt * 16 + l15) * 64 + kt * 32 + hi16;
        v16h bk = *(const v16h*)&KVs[base];   // B[k=d][n=s] = K[s][d]
        v16h bq = *(const v16h*)&Qs[base];    // B[k=d][n=s] = Q[s][d]
        s1 = wmma_f16(aq[kt], bk, s1);        // S   = Q_t · K_s^T
        s2 = wmma_f16(ak[kt], bq, s2);        // S^T = K_t · Q_s^T
      }
#pragma unroll
      for (int i = 0; i < 8; ++i) {
        int row  = i + hi8;
        int scol = c0 + nt * 16 + l15;
        int t    = t0 + row;
        float lg = (wstd * s1[i] + wrec * s2[i]) * scale +
                   wdisc * dbs[scol - sLo];
        bool ok  = (scol <= t) && (t - scol <= WIN);
        Ll[row * SMAX + (scol - sLo)] = ok ? lg : -1e30f;
      }
    }
  }
  __syncthreads();

  // ---- exact softmax per row; stream attn out non-temporally --------------
  for (int r = 0; r < 16; ++r) {
    float m = -3.0e38f;
    for (int j = lane; j < nS; j += 32) m = fmaxf(m, Ll[r * SMAX + j]);
#pragma unroll
    for (int off = 16; off > 0; off >>= 1) m = fmaxf(m, __shfl_xor(m, off, 32));
    float sum = 0.f;
    for (int j = lane; j < nS; j += 32) {
      float p = __expf(Ll[r * SMAX + j] - m);
      Ll[r * SMAX + j] = p;
      sum += p;
    }
#pragma unroll
    for (int off = 16; off > 0; off >>= 1) sum += __shfl_xor(sum, off, 32);
    float inv = 1.f / sum;
    float* arow = attn + ((size_t)h * T + (size_t)(t0 + r)) * T + sLo;
    for (int j = lane; j < nS; j += 32) {
      float p = Ll[r * SMAX + j] * inv;
      Pl[r * SMAX + j] = (_Float16)p;
      __builtin_nontemporal_store(p, &arow[j]);   // 805MB stream, bypass cache
    }
    for (int j = nS + lane; j < SMAX; j += 32)    // zero chunk-tail for PV frags
      Pl[r * SMAX + j] = (_Float16)0.f;
  }
  __syncthreads();

  // ---- ctx = P · V over the band ------------------------------------------
  v8f co[4];
#pragma unroll
  for (int nt = 0; nt < 4; ++nt)
    co[nt] = (v8f){0.f, 0.f, 0.f, 0.f, 0.f, 0.f, 0.f, 0.f};

  for (int c0 = sLo; c0 <= sHi; c0 += 32) {
    __syncthreads();
    // stage V chunk transposed: Vt[d][s_local] (reuse KVs); batch the loads
    v8h vv[8];
#pragma unroll
    for (int j = 0; j < 8; ++j) {
      int u = lane + j * 32;
      int r = u >> 3, q = u & 7;       // r = s_local, q*8 = d base
      vv[j] = *(const v8h*)&Vh[(size_t)(c0 + r) * HD + h * 64 + q * 8];
    }
#pragma unroll
    for (int j = 0; j < 8; ++j) {
      int u = lane + j * 32;
      int r = u >> 3, q = u & 7;
#pragma unroll
      for (int t = 0; t < 8; ++t) KVs[(q * 8 + t) * 32 + r] = vv[j][t];
    }
    __syncthreads();

    // A = P tile 16(t) x 32(s), contiguous f16 per lane
    const v8h* pp = (const v8h*)&Pl[l15 * SMAX + (c0 - sLo)];
    v16h ap = cat8(pp[2 * hi], pp[2 * hi + 1]);

#pragma unroll
    for (int nt = 0; nt < 4; ++nt) {
      v16h bv = *(const v16h*)&KVs[(nt * 16 + l15) * 32 + hi16];  // B[k=s][n=d]
      co[nt] = wmma_f16(ap, bv, co[nt]);
    }
  }

#pragma unroll
  for (int nt = 0; nt < 4; ++nt)
#pragma unroll
    for (int i = 0; i < 8; ++i)
      ctx[(size_t)(t0 + i + hi8) * HD + h * 64 + nt * 16 + l15] = co[nt][i];
}

// ---------------------------------------------------------------------------
// Launch
// ---------------------------------------------------------------------------
extern "C" void kernel_launch(void* const* d_in, const int* in_sizes, int n_in,
                              void* d_out, int out_size, void* d_ws, size_t ws_size,
                              hipStream_t stream) {
  (void)in_sizes; (void)n_in; (void)out_size; (void)ws_size;

  const float* x     = (const float*)d_in[0];   // [T,E]
  const float* Wq    = (const float*)d_in[1];   // [E,HD]
  const float* Wk    = (const float*)d_in[2];
  const float* Wv    = (const float*)d_in[3];
  const float* Wo    = (const float*)d_in[4];   // [HD,E]
  const float* gates = (const float*)d_in[5];   // [H,3]
  const float* u     = (const float*)d_in[6];   // [H,D]

  float* out  = (float*)d_out;                  // [T,E]
  float* attn = out + (size_t)T * E;            // [H,T,T]

  // workspace layout
  float*     Kp  = (float*)d_ws;                   // [T,HD] f32 (for dbias)
  float*     ctx = Kp + (size_t)T * HD;            // [T,HD] f32
  float*     db  = ctx + (size_t)T * HD;           // [H,T]
  _Float16*  Qh  = (_Float16*)(db + (size_t)H * T);// [T,HD] f16
  _Float16*  Kh  = Qh + (size_t)T * HD;
  _Float16*  Vh  = Kh + (size_t)T * HD;

  // 1) zero attn
  {
    long long n4 = (long long)H * T * T / 4;
    unsigned blocks = (unsigned)((n4 + 255) / 256);
    fill_zero_f4<<<dim3(blocks), 256, 0, stream>>>(attn, n4);
  }

  // 2) Q/K/V projections (f16 twins; f32 kept only for K)
  dim3 gg(HD / 64, T / 64);  // 12 x 64
  wmma_gemm64<<<gg, 128, 0, stream>>>(x, Wq, (float*)nullptr, Qh, T, HD, E);
  wmma_gemm64<<<gg, 128, 0, stream>>>(x, Wk, Kp, Kh, T, HD, E);
  wmma_gemm64<<<gg, 128, 0, stream>>>(x, Wv, (float*)nullptr, Vh, T, HD, E);

  // 3) per-head mean-centered K·u bias
  dbias_kernel<<<H, 256, 0, stream>>>(Kp, u, db);

  // 4) banded gated attention (writes attn + ctx)
  attn_kernel<<<dim3(T / 16, H), 32, 0, stream>>>(Qh, Kh, Vh, db, gates, attn, ctx);

  // 5) output projection
  dim3 go(E / 64, T / 64);   // 12 x 64
  wmma_gemm64<<<go, 128, 0, stream>>>(ctx, Wo, out, (_Float16*)nullptr, T, E, HD);
}
